// ArcticMoE_44650480009645
// MI455X (gfx1250) — matrix operand order repaired and checked by reading, not compiled, not run on hardware
//
#include <hip/hip_runtime.h>
#include <hip/hip_bf16.h>

// ---------------------------------------------------------------------------
// ArcticMoE top-2 MoE layer for MI455X (gfx1250), bf16 WMMA path.
//   T=4096 tokens, H=2048 hidden, I=2048 intermediate, E=8 experts, top-2.
// Sparse (top-2) FLOPs ~2.1e11; bytes ~450MB -> ~19us @ 23.3TB/s. Weights per
// expert (48MB) fit in the 192MB L2; with M=32 token tiles each expert's
// weights are re-read ~Ne/32 (~32) times, all L2 hits. Each converted bf16 B
// fragment is reused across two 16x16x32 WMMAs (two M-tiles).
// ---------------------------------------------------------------------------

constexpr int T  = 4096;
constexpr int H  = 2048;
constexpr int II = 2048;          // intermediate size (I)
constexpr int E  = 8;
constexpr int MT = 32;            // tokens per workgroup (two 16-row M-tiles)
constexpr int TILES = T / MT;     // 128 tiles per expert (worst case)

typedef __attribute__((ext_vector_type(16))) __bf16 v16bf;
typedef __attribute__((ext_vector_type(8)))  float  v8f;

union ABfrag {            // one WMMA bf16 A/B operand: 16 bf16 = 8 VGPRs
    v16bf v;
    uint4 q[2];           // two 16-byte halves for LDS b128 loads
};

__device__ __forceinline__ float fast_silu(float g) {
    // g * sigmoid(g) with v_rcp_f32 instead of the IEEE divide sequence
    return g * __builtin_amdgcn_rcpf(1.f + __expf(-g));
}

// ---------------------------------------------------------------------------
// Step 2: x fp32 -> bf16 (done once; A operand is reused heavily)
// ---------------------------------------------------------------------------
__global__ __launch_bounds__(256) void cvt_kernel(const float* __restrict__ x,
                                                  __hip_bfloat16* __restrict__ xb) {
    int i = (blockIdx.x * 256 + threadIdx.x) * 4;
    float4 v = *(const float4*)(x + i);
    __hip_bfloat16 o[4] = { __float2bfloat16(v.x), __float2bfloat16(v.y),
                            __float2bfloat16(v.z), __float2bfloat16(v.w) };
    *(uint2*)(xb + i) = *(const uint2*)o;
}

// ---------------------------------------------------------------------------
// Step 3: router. One wave32 per token: strided dot products over H for all 8
// experts, butterfly reduce, softmax + top-2 + renormalize, then compact
// (token, weight) into per-expert buckets with atomics.
// ---------------------------------------------------------------------------
__global__ __launch_bounds__(256) void router_kernel(const float* __restrict__ x,
                                                     const float* __restrict__ gw,
                                                     int* __restrict__ counts,
                                                     int* __restrict__ ids,
                                                     float* __restrict__ wts) {
    int t    = blockIdx.x * 8 + (threadIdx.x >> 5);   // 8 waves per block
    int lane = threadIdx.x & 31;
    if (t >= T) return;

    float acc[E];
#pragma unroll
    for (int e = 0; e < E; ++e) acc[e] = 0.f;

    const float* xr = x + (size_t)t * H;
    for (int j = lane; j < H; j += 32) {
        float xv = xr[j];
#pragma unroll
        for (int e = 0; e < E; ++e) acc[e] += xv * gw[e * H + j];
    }
#pragma unroll
    for (int e = 0; e < E; ++e) {
        for (int m = 16; m >= 1; m >>= 1) acc[e] += __shfl_xor(acc[e], m, 32);
    }

    if (lane == 0) {
        float mx = acc[0];
#pragma unroll
        for (int e = 1; e < E; ++e) mx = fmaxf(mx, acc[e]);
        float p[E];
#pragma unroll
        for (int e = 0; e < E; ++e) p[e] = __expf(acc[e] - mx);
        int b0 = 0, b1 = -1;
        float v0 = p[0], v1 = -1.f;
#pragma unroll
        for (int e = 1; e < E; ++e) {
            if (p[e] > v0) { v1 = v0; b1 = b0; v0 = p[e]; b0 = e; }
            else if (p[e] > v1) { v1 = p[e]; b1 = e; }
        }
        float inv = __builtin_amdgcn_rcpf(v0 + v1);
        int pos0 = atomicAdd(&counts[b0], 1);
        ids[b0 * T + pos0] = t;  wts[b0 * T + pos0] = v0 * inv;
        int pos1 = atomicAdd(&counts[b1], 1);
        ids[b1 * T + pos1] = t;  wts[b1 * T + pos1] = v1 * inv;
    }
}

// ---------------------------------------------------------------------------
// Step 4: fused expert MLP. One workgroup = (expert, 32-token tile).
//   GEMM1: [32,H] x [H,2I]^T via v_wmma_f32_16x16x32_bf16 (4 acc: 2 M x g/u)
//   SwiGLU on the f32 accumulators, h tile kept in LDS (bf16, 128KB)
//   GEMM2: [32,I] x [I,H]^T, scaled by routing weight, atomic-add into out.
// LDS: 128KB (x tile) + 128KB (h tile) + 256B -> fits CDNA5's 320KB WGP LDS.
// ---------------------------------------------------------------------------
__global__ __launch_bounds__(256) void moe_expert_kernel(
        const __hip_bfloat16* __restrict__ xb,
        const float* __restrict__ ws,     // [E, 2I, H]
        const float* __restrict__ w2s,    // [E, H, I]
        const int*   __restrict__ counts,
        const int*   __restrict__ ids,
        const float* __restrict__ wts,
        float*       __restrict__ out)    // [T, H]
{
    extern __shared__ char smem[];
    __hip_bfloat16* xt    = (__hip_bfloat16*)smem;                  // [MT][H]
    __hip_bfloat16* ht    = (__hip_bfloat16*)(smem + MT * H * 2);   // [MT][I]
    int*            s_tid = (int*)(smem + MT * H * 2 + MT * II * 2);
    float*          s_wt  = (float*)(smem + MT * H * 2 + MT * II * 2 + 128);

    const int e    = blockIdx.x / TILES;
    const int tile = blockIdx.x % TILES;
    const int Ne   = counts[e];
    if (tile * MT >= Ne) return;                 // uniform early exit

    if (threadIdx.x < MT) {
        int idx  = tile * MT + threadIdx.x;
        bool vld = idx < Ne;
        s_tid[threadIdx.x] = vld ? ids[e * T + idx] : 0;
        s_wt[threadIdx.x]  = vld ? wts[e * T + idx] : 0.f;   // pad rows add 0
    }
    __syncthreads();

    // Stage the MT token rows of x (bf16) into LDS, 16B chunks.
    {
        uint4* dst = (uint4*)xt;
        const int nchunks = MT * H / 8;          // 8 bf16 per uint4
        for (int c = threadIdx.x; c < nchunks; c += 256) {
            int r   = c >> 8;                    // H/8 = 256 chunks per row
            int col = (c & 255) * 8;
            dst[c] = *(const uint4*)(xb + (size_t)s_tid[r] * H + col);
        }
    }
    __syncthreads();

    const int lane = threadIdx.x & 31;
    const int wave = threadIdx.x >> 5;
    const int half = (lane < 16) ? 0 : 1;        // lane-half of the wave
    const int mrow = lane & 15;                  // M row / N column index

    // ---------------- GEMM1 + SwiGLU ----------------
    const float* wsE = ws + (size_t)e * (2 * II) * H;
    for (int nt = wave; nt < II / 16; nt += 8) {
        const int    o    = nt * 16 + mrow;      // this lane's B column
        const float* gRow = wsE + (size_t)o * H;         // gate weight row
        const float* uRow = wsE + (size_t)(o + II) * H;  // up weight row
        v8f accG0 = {}, accU0 = {}, accG1 = {}, accU1 = {};
        for (int k0 = 0; k0 < H; k0 += 32) {
            // A fragments for the two 16-row M-tiles (bf16 A 16x32 layout)
            const int kA = k0 + half * 8;
            ABfrag a0, a1;
            a0.q[0] = *(const uint4*)(xt + (mrow)      * H + kA);
            a0.q[1] = *(const uint4*)(xt + (mrow)      * H + kA + 16);
            a1.q[0] = *(const uint4*)(xt + (16 + mrow) * H + kA);
            a1.q[1] = *(const uint4*)(xt + (16 + mrow) * H + kA + 16);
            // B fragments: K = 8h+0..7 and 16+8h+0..7 of one column, fp32->bf16
            const int kB = k0 + half * 8;
            __builtin_prefetch(gRow + kB + 128, 0, 0);   // global_prefetch_b8
            ABfrag bg, bu;
#pragma unroll
            for (int jj = 0; jj < 4; ++jj) {
                float4 fg = *(const float4*)(gRow + kB + jj * 4 + (jj & 2) * 4);
                float4 fu = *(const float4*)(uRow + kB + jj * 4 + (jj & 2) * 4);
                bg.v[jj * 4 + 0] = (__bf16)fg.x; bg.v[jj * 4 + 1] = (__bf16)fg.y;
                bg.v[jj * 4 + 2] = (__bf16)fg.z; bg.v[jj * 4 + 3] = (__bf16)fg.w;
                bu.v[jj * 4 + 0] = (__bf16)fu.x; bu.v[jj * 4 + 1] = (__bf16)fu.y;
                bu.v[jj * 4 + 2] = (__bf16)fu.z; bu.v[jj * 4 + 3] = (__bf16)fu.w;
            }
            accG0 = __builtin_amdgcn_wmma_f32_16x16x32_bf16(
                        false, a0.v, false, bg.v, (short)0, accG0, false, false);
            accU0 = __builtin_amdgcn_wmma_f32_16x16x32_bf16(
                        false, a0.v, false, bu.v, (short)0, accU0, false, false);
            accG1 = __builtin_amdgcn_wmma_f32_16x16x32_bf16(
                        false, a1.v, false, bg.v, (short)0, accG1, false, false);
            accU1 = __builtin_amdgcn_wmma_f32_16x16x32_bf16(
                        false, a1.v, false, bu.v, (short)0, accU1, false, false);
        }
        // h = silu(gate) * up, stored bf16 into the LDS h-tile
#pragma unroll
        for (int r = 0; r < 8; ++r) {
            int m = r + half * 8;                // C/D layout: M = r (+8)
            ht[(m)      * II + nt * 16 + mrow] =
                __float2bfloat16(fast_silu(accG0[r]) * accU0[r]);
            ht[(16 + m) * II + nt * 16 + mrow] =
                __float2bfloat16(fast_silu(accG1[r]) * accU1[r]);
        }
    }
    __syncthreads();

    // ---------------- GEMM2 + combine ----------------
    const float* w2E = w2s + (size_t)e * H * II;
    for (int nt = wave; nt < H / 16; nt += 8) {
        const int    hh   = nt * 16 + mrow;
        const float* dRow = w2E + (size_t)hh * II;
        v8f acc0 = {}, acc1 = {};
        for (int k0 = 0; k0 < II; k0 += 32) {
            const int kA = k0 + half * 8;
            ABfrag a0, a1;
            a0.q[0] = *(const uint4*)(ht + (mrow)      * II + kA);
            a0.q[1] = *(const uint4*)(ht + (mrow)      * II + kA + 16);
            a1.q[0] = *(const uint4*)(ht + (16 + mrow) * II + kA);
            a1.q[1] = *(const uint4*)(ht + (16 + mrow) * II + kA + 16);
            const int kB = k0 + half * 8;
            __builtin_prefetch(dRow + kB + 128, 0, 0);
            ABfrag b;
#pragma unroll
            for (int jj = 0; jj < 4; ++jj) {
                float4 f = *(const float4*)(dRow + kB + jj * 4 + (jj & 2) * 4);
                b.v[jj * 4 + 0] = (__bf16)f.x; b.v[jj * 4 + 1] = (__bf16)f.y;
                b.v[jj * 4 + 2] = (__bf16)f.z; b.v[jj * 4 + 3] = (__bf16)f.w;
            }
            acc0 = __builtin_amdgcn_wmma_f32_16x16x32_bf16(
                       false, a0.v, false, b.v, (short)0, acc0, false, false);
            acc1 = __builtin_amdgcn_wmma_f32_16x16x32_bf16(
                       false, a1.v, false, b.v, (short)0, acc1, false, false);
        }
#pragma unroll
        for (int r = 0; r < 8; ++r) {
            int m = r + half * 8;
            atomicAdd(out + (size_t)s_tid[m]      * H + nt * 16 + mrow,
                      acc0[r] * s_wt[m]);
            atomicAdd(out + (size_t)s_tid[16 + m] * H + nt * 16 + mrow,
                      acc1[r] * s_wt[16 + m]);
        }
    }
}

// ---------------------------------------------------------------------------
// Launch. Workspace layout (needs ~16.6 MB):
//   [0,64)                        expert counters (8 ints, zeroed per call)
//   [64, 64+E*T*4)                compacted token ids
//   [64+E*T*4, 64+E*T*8)          compacted routing weights
//   [64+E*T*8, ...)               x in bf16 (T*H*2 bytes, 16B aligned)
// ---------------------------------------------------------------------------
extern "C" void kernel_launch(void* const* d_in, const int* in_sizes, int n_in,
                              void* d_out, int out_size, void* d_ws, size_t ws_size,
                              hipStream_t stream) {
    const float* x   = (const float*)d_in[0];
    const float* gw  = (const float*)d_in[1];
    const float* ws  = (const float*)d_in[2];
    const float* w2s = (const float*)d_in[3];
    float*       out = (float*)d_out;

    char* wsp = (char*)d_ws;
    int*   counts = (int*)wsp;
    int*   ids    = (int*)(wsp + 64);
    float* wts    = (float*)(wsp + 64 + (size_t)E * T * 4);
    __hip_bfloat16* xb = (__hip_bfloat16*)(wsp + 64 + (size_t)E * T * 8);

    hipMemsetAsync(counts, 0, 64, stream);
    hipMemsetAsync(d_out, 0, (size_t)out_size * sizeof(float), stream);

    cvt_kernel<<<(T * H) / 1024, 256, 0, stream>>>(x, xb);
    router_kernel<<<T / 8, 256, 0, stream>>>(x, gw, counts, ids, wts);

    size_t shmem = (size_t)MT * H * 2 + (size_t)MT * II * 2 + 256;  // 256KB + ids/wts
    moe_expert_kernel<<<E * TILES, 256, shmem, stream>>>(
        xb, ws, w2s, counts, ids, wts, out);
}
